// ViscoelasticMaterialModelM_11484742549884
// MI455X (gfx1250) — matrix operand order repaired and checked by reading, not compiled, not run on hardware
//
#include <hip/hip_runtime.h>

// Viscoelastic scan for MI455X (gfx1250, wave32).
// Per time step the two grad-MLPs collapse to
//   z = c + u*a + v*b ; grads = (G1.relu(z), G2.relu(z))   (HID=128)
// One wave = 16 batch elements. relu(z) is a 128x16 f16 B-fragment built
// with v_pk_fma_f16/v_pk_max; the four gradient reductions are a chain of
// 4x v_wmma_f32_16x16x32_f16 against a constant G^T A-fragment (rows 0..3).
// D-layout leaves per-lane scalars; ds_bpermute broadcasts across lane groups.
// v1->v2: a/b coefficient fragments moved to LDS (wave-invariant, broadcast
// reads) to get under 256 VGPRs and kill the s_set_vgpr_msb storm; guarded
// stores replaced by duplicate full-wave stores to remove exec branches.

#define B_SZ   8192
#define T_SZ   512
#define EDIM   64
#define ENC    32
#define MFEAT  64
#define HID    128
#define DT_C   0.01f

typedef _Float16 h2   __attribute__((ext_vector_type(2)));
typedef _Float16 v16h __attribute__((ext_vector_type(16)));
typedef float    v8f  __attribute__((ext_vector_type(8)));

union V16H { v16h v; h2 h[8]; };
union U32B { uint4 q[2]; h2 h[8]; };   // one 32-byte coefficient chunk

__device__ __forceinline__ h2 make_h2(float a, float b) {
    h2 r = {(_Float16)a, (_Float16)b};
    return r;
}

// ---------------------------------------------------------------------------
// Kernel 1: microstructure features m[b,k] = concat(E@WE+bE, nu@Wnu+bnu)
// ---------------------------------------------------------------------------
__global__ void precompute_m(const float* __restrict__ E, const float* __restrict__ nu,
                             const float* __restrict__ WE, const float* __restrict__ bE,
                             const float* __restrict__ Wnu, const float* __restrict__ bnu,
                             float* __restrict__ m) {
    int idx = blockIdx.x * blockDim.x + threadIdx.x;
    if (idx >= B_SZ * MFEAT) return;
    int b = idx / MFEAT, k = idx % MFEAT;
    float acc;
    if (k < ENC) {
        acc = bE[k];
        const float* Er = E + (size_t)b * EDIM;
        #pragma unroll 8
        for (int d = 0; d < EDIM; ++d) acc += Er[d] * WE[d * ENC + k];
    } else {
        int kk = k - ENC;
        acc = bnu[kk];
        const float* Nr = nu + (size_t)b * EDIM;
        #pragma unroll 8
        for (int d = 0; d < EDIM; ++d) acc += Nr[d] * Wnu[d * ENC + kk];
    }
    m[idx] = acc;
}

// ---------------------------------------------------------------------------
// Kernel 2: time-invariant pre-activations c[b,K] = b1[K] + m[b,:].W1[2:,K],
// stored f16 in wave32 WMMA B-fragment order:
//   half slot s (0..63) of lane L of tile wt:
//   ch=s/16, rr=(s%16)/2, h=s&1, g=L/16, b=wt*16+L%16, K=32ch+16g+2rr+h
// ---------------------------------------------------------------------------
__global__ void precompute_c(const float* __restrict__ m,
                             const float* __restrict__ W1e, const float* __restrict__ b1e,
                             const float* __restrict__ W1d, const float* __restrict__ b1d,
                             _Float16* __restrict__ c_d, _Float16* __restrict__ c_e) {
    const int per = (B_SZ / 16) * 32 * 64;   // == B_SZ*HID
    int idx = blockIdx.x * blockDim.x + threadIdx.x;
    if (idx >= 2 * per) return;
    int mlp = idx / per;                      // 0 = dissipation, 1 = energy
    int t   = idx % per;
    int wt  = t / (32 * 64);
    int L   = (t / 64) % 32;
    int s   = t % 64;
    int ch  = s / 16;
    int rr  = (s % 16) / 2;
    int h   = s & 1;
    int g   = L / 16;
    int b   = wt * 16 + (L % 16);
    int K   = 32 * ch + 16 * g + 2 * rr + h;

    const float* W1 = mlp ? W1e : W1d;
    const float* b1 = mlp ? b1e : b1d;
    float acc = b1[K];
    const float* mr = m + (size_t)b * MFEAT;
    #pragma unroll 8
    for (int k = 0; k < MFEAT; ++k) acc += mr[k] * W1[(2 + k) * HID + K];
    if (mlp) c_e[t] = (_Float16)acc;
    else     c_d[t] = (_Float16)acc;
}

// ---------------------------------------------------------------------------
// Kernel 3: sequential scan. One wave32 = one 16-batch tile.
// ---------------------------------------------------------------------------
__global__ __launch_bounds__(256, 1)
void scan_kernel(const float* __restrict__ e, const float* __restrict__ ed,
                 const float* __restrict__ W1e, const float* __restrict__ W2e,
                 const float* __restrict__ W1d, const float* __restrict__ W2d,
                 const _Float16* __restrict__ c_d_g, const _Float16* __restrict__ c_e_g,
                 float* __restrict__ stress_out, float* __restrict__ xi_out) {
    const int lane = threadIdx.x & 31;
    const int wt   = blockIdx.x * (blockDim.x >> 5) + (threadIdx.x >> 5);
    const int g    = lane >> 4;        // lane group: K offset +16*g in B-fragments
    const int M    = lane & 15;        // batch column (B/D) / row (A) of this lane
    const int b    = wt * 16 + M;

    // ---- wave-invariant u/v coefficient table in LDS ----
    // scoef[vec][g][ch][rr] : vec 0=W1d row0(u), 1=W1d row1(v), 2=W1e row0, 3=W1e row1
    // Each (vec,g,ch) row is 32B; all lanes of a group read the same address
    // (LDS broadcast) via ds_load_b128.
    __shared__ __align__(16) h2 scoef[4][2][4][8];
    {
        int t = threadIdx.x;           // 256 threads -> 256 entries
        int vec = t >> 6, gg = (t >> 5) & 1, ch = (t >> 3) & 3, rr = t & 7;
        int K0 = 32 * ch + 16 * gg + 2 * rr;
        const float* W1 = (vec < 2) ? W1d : W1e;
        int row = vec & 1;
        scoef[vec][gg][ch][rr] = make_h2(W1[row * HID + K0], W1[row * HID + K0 + 1]);
    }
    __syncthreads();

    // ---- per-lane time-invariant c fragments (f16, fragment-ordered) ----
    h2 cd[32], ce[32];
    {
        const h2* pd = (const h2*)(c_d_g + ((size_t)wt * 32 + lane) * 64);
        const h2* pe = (const h2*)(c_e_g + ((size_t)wt * 32 + lane) * 64);
        #pragma unroll
        for (int r = 0; r < 32; ++r) { cd[r] = pd[r]; ce[r] = pe[r]; }
    }

    // ---- constant A-fragment: G^T (16x128), rows 0..3 used, rest zero.
    // A fragment k order used here (contiguous, same as B): lane row = M; reg rr, half h ->
    //   K = 32*ch + 16*g + 2*rr + h
    V16H gt[4];
    #pragma unroll
    for (int ch = 0; ch < 4; ++ch) {
        #pragma unroll
        for (int rr = 0; rr < 8; ++rr) {
            int Kb = 32 * ch + 16 * g + 2 * rr;
            float v0 = 0.f, v1 = 0.f;
            if (M == 0)      { v0 = 2.f * W2d[Kb] * W1d[Kb];           v1 = 2.f * W2d[Kb+1] * W1d[Kb+1]; }
            else if (M == 1) { v0 = 2.f * W2d[Kb] * W1d[HID + Kb];     v1 = 2.f * W2d[Kb+1] * W1d[HID + Kb+1]; }
            else if (M == 2) { v0 = 2.f * W2e[Kb] * W1e[Kb];           v1 = 2.f * W2e[Kb+1] * W1e[Kb+1]; }
            else if (M == 3) { v0 = 2.f * W2e[Kb] * W1e[HID + Kb];     v1 = 2.f * W2e[Kb+1] * W1e[HID + Kb+1]; }
            gt[ch].h[rr] = make_h2(v0, v1);
        }
    }

    const h2 hzero = make_h2(0.f, 0.f);

    // One MLP gradient evaluation: z = c + u*a + v*b, relu, 4-chunk WMMA chain.
    // vecbase: 0 = dissipation MLP, 2 = energy MLP.
    auto phase = [&](float u, float v, const h2* c, int vecbase) -> v8f {
        h2 uu = make_h2(u, u);
        h2 vv = make_h2(v, v);
        v8f acc = {};
        #pragma unroll
        for (int ch = 0; ch < 4; ++ch) {
            U32B au, bv;
            {
                const uint4* pa = (const uint4*)&scoef[vecbase][g][ch][0];
                const uint4* pb = (const uint4*)&scoef[vecbase + 1][g][ch][0];
                au.q[0] = pa[0]; au.q[1] = pa[1];   // ds_load_b128 x2 (broadcast)
                bv.q[0] = pb[0]; bv.q[1] = pb[1];
            }
            V16H zb;
            #pragma unroll
            for (int rr = 0; rr < 8; ++rr) {
                h2 z = au.h[rr] * uu + c[ch * 8 + rr];   // v_pk_fma_f16
                z = bv.h[rr] * vv + z;                   // v_pk_fma_f16
                zb.h[rr] = __builtin_elementwise_max(z, hzero);  // v_pk_max
            }
            acc = __builtin_amdgcn_wmma_f32_16x16x32_f16(
                false, gt[ch].v, false, zb.v, (short)0, acc, false, false);
        }
        return acc;
    };

    // broadcast lane M's value (valid in lanes 0..15) to both lane groups
    auto bcast = [&](float x) -> float {
        int xi2 = __builtin_bit_cast(int, x);
        int r = __builtin_amdgcn_ds_bpermute(M << 2, xi2);
        return __builtin_bit_cast(float, r);
    };

    float xi = 0.f;
    const size_t rowB = (size_t)b * T_SZ;

    // initial energy gradients at (e[:,0], xi=0)
    float u0 = e[rowB + 0];
    v8f De = phase(u0, 0.f, ce, 2);
    float s_eq = bcast(De[2]);
    float dcur = bcast(De[3]);
    xi_out[rowB + 0] = 0.f;                       // lanes M and M+16 duplicate

    float s_neq = 0.f;
    for (int i = 0; i < T_SZ - 1; ++i) {
        // dissipation gradients at (e_dot[:,i], -d)
        float up = ed[rowB + i];
        v8f Dd = phase(up, -dcur, cd, 0);
        s_neq     = bcast(Dd[0]);
        float kin = bcast(Dd[1]);

        xi += DT_C * kin;
        stress_out[rowB + i] = s_eq - s_neq;      // duplicate identical stores
        xi_out[rowB + i + 1] = xi;

        // energy gradients at (e[:,i+1], xi_n)
        float ui = e[rowB + i + 1];
        De   = phase(ui, xi, ce, 2);
        s_eq = bcast(De[2]);
        dcur = bcast(De[3]);
    }
    stress_out[rowB + (T_SZ - 1)] = s_eq - s_neq;
}

// ---------------------------------------------------------------------------
extern "C" void kernel_launch(void* const* d_in, const int* in_sizes, int n_in,
                              void* d_out, int out_size, void* d_ws, size_t ws_size,
                              hipStream_t stream) {
    (void)in_sizes; (void)n_in; (void)out_size; (void)ws_size;

    const float* e   = (const float*)d_in[0];
    const float* ed  = (const float*)d_in[1];
    const float* E   = (const float*)d_in[2];
    const float* nu  = (const float*)d_in[3];
    const float* W1e = (const float*)d_in[4];
    const float* b1e = (const float*)d_in[5];
    const float* W2e = (const float*)d_in[6];
    /* b2e = d_in[7] : bias drops out of the gradients */
    const float* W1d = (const float*)d_in[8];
    const float* b1d = (const float*)d_in[9];
    const float* W2d = (const float*)d_in[10];
    /* b2d = d_in[11] : unused */
    const float* WE  = (const float*)d_in[12];
    const float* bE  = (const float*)d_in[13];
    const float* Wnu = (const float*)d_in[14];
    const float* bnu = (const float*)d_in[15];

    // workspace: m (B*64 f32 = 2MB) | c_d (B*128 f16 = 2MB) | c_e (2MB)
    float*    m   = (float*)d_ws;
    _Float16* c_d = (_Float16*)((char*)d_ws + (size_t)B_SZ * MFEAT * sizeof(float));
    _Float16* c_e = c_d + (size_t)B_SZ * HID;

    float* stress_out = (float*)d_out;
    float* xi_out     = stress_out + (size_t)B_SZ * T_SZ;

    precompute_m<<<(B_SZ * MFEAT + 255) / 256, 256, 0, stream>>>(E, nu, WE, bE, Wnu, bnu, m);
    precompute_c<<<(2 * B_SZ * HID + 255) / 256, 256, 0, stream>>>(m, W1e, b1e, W1d, b1d, c_d, c_e);

    // 512 wave-tiles (B/16), 8 waves per 256-thread block -> 64 blocks
    scan_kernel<<<(B_SZ / 16) / 8, 256, 0, stream>>>(
        e, ed, W1e, W2e, W1d, W2d, c_d, c_e, stress_out, xi_out);
}
